// ContrastiveLoss_dot_OT_38448547234380
// MI455X (gfx1250) — matrix operand order repaired and checked by reading, not compiled, not run on hardware
//
#include <hip/hip_runtime.h>
#include <stdint.h>

// ---- CDNA5 WMMA types (wave32) ----
typedef __bf16 v16bf __attribute__((ext_vector_type(16)));
typedef float  v8f   __attribute__((ext_vector_type(8)));

#define N_IMAGE   128
#define N_REG     36
#define DDIM      1024
#define MAX_WORD  60
#define SPAD_W    64                       // words padded to 64 (4 N-tiles)
#define IM_ROWS   (N_IMAGE * N_REG)        // 4608
#define IM_ROWS_PAD 4624                   // pad: last block tiles reach row 4615
#define S_ROWS_PAD  (N_IMAGE * SPAD_W)     // 8192
#define MARGIN_C  0.2f
#define ALPHA_C   0.1f
#define OT_ITER_C 20
#define EPSN      1e-12f
#define MS        65                       // LDS row stride (bank-conflict pad)

union FragU { uint4 u4[2]; v16bf v; };

__device__ inline unsigned short f2bf(float f) {
  union { float f; unsigned u; } c; c.f = f;
  unsigned x = c.u;
  unsigned r = x + 0x7fffu + ((x >> 16) & 1u);   // RNE
  return (unsigned short)(r >> 16);
}

// ---------- Kernel 1: row norms + bf16 conversion (zero-pads tiles) ----------
__global__ void prep_kernel(const float* __restrict__ im, const float* __restrict__ s,
                            unsigned short* __restrict__ imbf, unsigned short* __restrict__ sbf,
                            float* __restrict__ nim, float* __restrict__ ncap) {
  const int t = threadIdx.x;               // 128 threads, 8 elems each
  const int bid = blockIdx.x;
  __shared__ float red[128];
  const float* src = nullptr;
  unsigned short* dst;
  float* normOut = nullptr;
  if (bid < IM_ROWS_PAD) {
    dst = imbf + (size_t)bid * DDIM;
    if (bid < IM_ROWS) { src = im + (size_t)bid * DDIM; normOut = nim + bid; }
  } else {
    const int idx = bid - IM_ROWS_PAD;
    const int i = idx >> 6, w = idx & 63;
    dst = sbf + (size_t)idx * DDIM;
    if (w < MAX_WORD) { src = s + ((size_t)i * MAX_WORD + w) * DDIM; normOut = ncap + i * MAX_WORD + w; }
  }
  float v[8]; float ss = 0.f;
  if (src) {
    #pragma unroll
    for (int e = 0; e < 8; ++e) { v[e] = src[t * 8 + e]; ss += v[e] * v[e]; }
  } else {
    #pragma unroll
    for (int e = 0; e < 8; ++e) v[e] = 0.f;
  }
  union { unsigned short h[8]; uint4 u; } pk;
  #pragma unroll
  for (int e = 0; e < 8; ++e) pk.h[e] = f2bf(v[e]);
  *(uint4*)(dst + t * 8) = pk.u;
  red[t] = ss;
  __syncthreads();
  for (int sd = 64; sd > 0; sd >>= 1) { if (t < sd) red[t] += red[t + sd]; __syncthreads(); }
  if (t == 0 && normOut) *normOut = sqrtf(red[0]);
}

// K-loop over NT active 16-col N-tiles for this wave (NT is wave-uniform).
template <int NT>
__device__ inline void gemm_tiles(const unsigned short* __restrict__ imbf,
                                  const unsigned short* __restrict__ sbf,
                                  size_t arow0, size_t brow0,
                                  int lrow, int lhalf, v8f acc[5][2]) {
  for (int k0 = 0; k0 < DDIM; k0 += 32) {
    FragU bfr[NT];
    #pragma unroll
    for (int n = 0; n < NT; ++n) {
      // B (32x16): lanes 0-15 -> col=lane, K=k0..k0+15 ; lanes 16-31 -> K=k0+16..k0+31
      const unsigned short* bp = sbf + (brow0 + n * 16 + lrow) * DDIM + k0 + lhalf * 16;
      bfr[n].u4[0] = *(const uint4*)(bp);
      bfr[n].u4[1] = *(const uint4*)(bp + 8);
    }
    if (k0 + 32 < DDIM)
      __builtin_prefetch(sbf + (brow0 + lrow) * DDIM + k0 + 32, 0, 1); // global_prefetch_b8
    #pragma unroll
    for (int mt = 0; mt < 5; ++mt) {
      // A (16x32): lanes 0-15 -> K 0..15; lanes 16-31 -> K 16..31
      const unsigned short* ap = imbf + (arow0 + mt * 16 + lrow) * DDIM + k0 + lhalf * 16;
      FragU af;
      af.u4[0] = *(const uint4*)(ap);
      af.u4[1] = *(const uint4*)(ap + 8);
      #pragma unroll
      for (int n = 0; n < NT; ++n)
        acc[mt][n] = __builtin_amdgcn_wmma_f32_16x16x32_bf16(
            false, af.v, false, bfr[n].v, (short)0, acc[mt][n], false, false);
    }
  }
}

// ---------- Kernel 2: WMMA GEMM slab + sims + IPOT, per (caption, 2 images) ----------
__global__ void __launch_bounds__(64) score_kernel(
    const unsigned short* __restrict__ imbf, const unsigned short* __restrict__ sbf,
    const float* __restrict__ nim, const float* __restrict__ ncap,
    const int* __restrict__ lens,
    float* __restrict__ scores, float* __restrict__ scores_ot) {

  __shared__ float Mlds[80 * MS];          // GEMM result; later A = exp(-2C) (rows [0,72))
  __shared__ float Tlds[72 * MS];          // IPOT transport plan
  __shared__ float sig[2][64];
  __shared__ float del[2][36];

  const int lane  = threadIdx.x & 31;
  const int wid   = threadIdx.x >> 5;      // 0..1: N-tile pair owner / local image b
  const int i     = blockIdx.x;            // caption index
  const int b0    = blockIdx.y * 2;        // first image of this block
  int m = lens[i]; m = m < 1 ? 1 : (m > MAX_WORD ? MAX_WORD : m);
  const int ntU   = (m + 15) >> 4;         // live N-tiles in block: 1..4 (uniform)
  int active = ntU - wid * 2;              // live N-tiles for this wave: 0..2
  active = active < 0 ? 0 : (active > 2 ? 2 : active);

  const int lrow  = lane & 15;
  const int lhalf = lane >> 4;

  // ---- bf16 WMMA GEMM: M[(b,r), w] = im[b,r,:] . s[i,w,:] ----
  v8f acc[5][2];
  #pragma unroll
  for (int mt = 0; mt < 5; ++mt)
    #pragma unroll
    for (int n = 0; n < 2; ++n)
      #pragma unroll
      for (int e = 0; e < 8; ++e) acc[mt][n][e] = 0.f;

  const size_t arow0 = (size_t)b0 * N_REG;             // 72 rows, tiles cover 80 (padded)
  const size_t brow0 = (size_t)i * SPAD_W + wid * 32;  // this wave's 2 N-tiles

  if (active == 2)      gemm_tiles<2>(imbf, sbf, arow0, brow0, lrow, lhalf, acc);
  else if (active == 1) gemm_tiles<1>(imbf, sbf, arow0, brow0, lrow, lhalf, acc);

  // C/D scatter: VGPR g -> row g + 8*(lane>=16), col lane&15
  for (int n = 0; n < active; ++n)
    #pragma unroll
    for (int mt = 0; mt < 5; ++mt)
      #pragma unroll
      for (int g = 0; g < 8; ++g)
        Mlds[(mt * 16 + g + lhalf * 8) * MS + wid * 32 + n * 16 + lrow] = acc[mt][n][g];
  __syncthreads();

  // ---- sims (MISA): mean_w max_r M ----
  {
    float ssum = 0.f;
    for (int w = lane; w < m; w += 32) {
      float mx = -3.4e38f;
      #pragma unroll 4
      for (int r = 0; r < N_REG; ++r)
        mx = fmaxf(mx, Mlds[(wid * N_REG + r) * MS + w]);
      ssum += mx;
    }
    #pragma unroll
    for (int off = 16; off > 0; off >>= 1) ssum += __shfl_down(ssum, off, 32);
    if (lane == 0) scores[(size_t)(b0 + wid) * N_IMAGE + i] = ssum / (float)m;
  }
  __syncthreads();

  // ---- fold norms: C = 1 - M/((|im|+eps)(|cap|+eps)); store A=exp(-2C) in place ----
  for (int r = 0; r < N_REG; ++r) {
    const float invn = 1.f / (nim[(size_t)(b0 + wid) * N_REG + r] + EPSN);
    for (int w = lane; w < m; w += 32) {
      const float c = 1.f - Mlds[(wid * N_REG + r) * MS + w] * invn /
                            (ncap[(size_t)i * MAX_WORD + w] + EPSN);
      Mlds[(wid * N_REG + r) * MS + w] = __expf(-2.0f * c);  // A ; C = -0.5*ln(A)
      Tlds[(wid * N_REG + r) * MS + w] = 1.f;
    }
  }
  for (int w = lane; w < m; w += 32) sig[wid][w] = 1.f / (float)m;
  __syncthreads();

  // ---- IPOT, 20 iterations; one wave per image b ----
  const float invn36 = 1.f / 36.f;
  for (int it = 0; it < OT_ITER_C; ++it) {
    for (int r = lane; r < N_REG; r += 32) {         // delta = 1/(n * Q sigma)
      float a = 0.f;
      for (int w = 0; w < m; ++w)
        a += Mlds[(wid * N_REG + r) * MS + w] * Tlds[(wid * N_REG + r) * MS + w] * sig[wid][w];
      del[wid][r] = invn36 / a;
    }
    __syncthreads();
    for (int w = lane; w < m; w += 32) {             // sigma = 1/(m * Q^T delta)
      float a = 0.f;
      for (int r = 0; r < N_REG; ++r)
        a += Mlds[(wid * N_REG + r) * MS + w] * Tlds[(wid * N_REG + r) * MS + w] * del[wid][r];
      sig[wid][w] = 1.f / ((float)m * a);
    }
    __syncthreads();
    for (int r = 0; r < N_REG; ++r) {                // T = delta * (A*T) * sigma^T
      const float dr = del[wid][r];
      for (int w = lane; w < m; w += 32)
        Tlds[(wid * N_REG + r) * MS + w] =
            dr * Mlds[(wid * N_REG + r) * MS + w] * Tlds[(wid * N_REG + r) * MS + w] * sig[wid][w];
    }
    __syncthreads();
  }

  // ---- distance = -sum(C * T), C recovered as -0.5*ln(A) ----
  {
    float part = 0.f;
    for (int r = 0; r < N_REG; ++r)
      for (int w = lane; w < m; w += 32)
        part += (-0.5f * __logf(Mlds[(wid * N_REG + r) * MS + w])) *
                Tlds[(wid * N_REG + r) * MS + w];
    #pragma unroll
    for (int off = 16; off > 0; off >>= 1) part += __shfl_down(part, off, 32);
    if (lane == 0) scores_ot[(size_t)(b0 + wid) * N_IMAGE + i] = -part;
  }
}

// ---------- Kernel 3: margin loss with max-violation ----------
__global__ void loss_kernel(const float* __restrict__ sc, const float* __restrict__ sot,
                            float* __restrict__ out) {
  const int t = threadIdx.x;               // 0..127 : row t and column t
  __shared__ float red0[128], red1[128];
  const float dx  = sc[t * N_IMAGE + t];
  const float dxo = sot[t * N_IMAGE + t];
  float rmax = 0.f, rmax_o = 0.f, cmax = 0.f, cmax_o = 0.f;
  for (int j = 0; j < N_IMAGE; ++j) {
    if (j == t) continue;
    const float v  = fmaxf(0.f, MARGIN_C + sc[t * N_IMAGE + j] - dx);   // cost_s row t
    const float vo = fmaxf(0.f, MARGIN_C + sot[t * N_IMAGE + j] - dxo);
    rmax   = fmaxf(rmax, v + ALPHA_C * vo);
    rmax_o = fmaxf(rmax_o, vo);
    const float w  = fmaxf(0.f, MARGIN_C + sc[j * N_IMAGE + t] - dx);   // cost_im col t
    const float wo = fmaxf(0.f, MARGIN_C + sot[j * N_IMAGE + t] - dxo);
    cmax   = fmaxf(cmax, w + ALPHA_C * wo);
    cmax_o = fmaxf(cmax_o, wo);
  }
  red0[t] = rmax + cmax; red1[t] = rmax_o + cmax_o;
  __syncthreads();
  for (int s = 64; s > 0; s >>= 1) {
    if (t < s) { red0[t] += red0[t + s]; red1[t] += red1[t + s]; }
    __syncthreads();
  }
  if (t == 0) { out[0] = red0[0]; out[1] = red1[0]; }
}

extern "C" void kernel_launch(void* const* d_in, const int* in_sizes, int n_in,
                              void* d_out, int out_size, void* d_ws, size_t ws_size,
                              hipStream_t stream) {
  (void)in_sizes; (void)n_in; (void)out_size; (void)ws_size;
  const float* im = (const float*)d_in[0];
  const float* s  = (const float*)d_in[1];
  const int*  lens = (const int*)d_in[2];   // cap lengths (10..60)
  float* out = (float*)d_out;

  char* ws = (char*)d_ws;
  size_t off = 0;
  unsigned short* imbf = (unsigned short*)(ws + off); off += (size_t)IM_ROWS_PAD * DDIM * 2;
  unsigned short* sbf  = (unsigned short*)(ws + off); off += (size_t)S_ROWS_PAD  * DDIM * 2;
  float* nim       = (float*)(ws + off); off += (size_t)IM_ROWS * 4;
  float* ncap      = (float*)(ws + off); off += (size_t)N_IMAGE * MAX_WORD * 4;
  float* scores    = (float*)(ws + off); off += (size_t)N_IMAGE * N_IMAGE * 4;
  float* scores_ot = (float*)(ws + off); off += (size_t)N_IMAGE * N_IMAGE * 4;

  prep_kernel<<<IM_ROWS_PAD + S_ROWS_PAD, 128, 0, stream>>>(im, s, imbf, sbf, nim, ncap);
  score_kernel<<<dim3(N_IMAGE, N_IMAGE / 2), 64, 0, stream>>>(imbf, sbf, nim, ncap, lens,
                                                              scores, scores_ot);
  loss_kernel<<<1, 128, 0, stream>>>(scores, scores_ot, out);
}
